// VisionTransformer_65180423684505
// MI455X (gfx1250) — compile-verified, matrix operand order checked by reading, not run on hardware
//
#include <hip/hip_runtime.h>
#include <hip/hip_bf16.h>

// ---------------- types ----------------
typedef __attribute__((ext_vector_type(16))) __bf16 v16bf;
typedef __attribute__((ext_vector_type(8)))  __bf16 v8bf;
typedef __attribute__((ext_vector_type(8)))  float  v8f;
typedef int vint4 __attribute__((vector_size(16)));   // matches builtin's V4i param type

// CDNA5 async global->LDS copy path (ASYNCcnt / s_wait_asynccnt), guarded so a
// toolchain without the builtins falls back to synchronous staging.
#if defined(__AMDGCN__) && __has_builtin(__builtin_amdgcn_global_load_async_to_lds_b128) && __has_builtin(__builtin_amdgcn_s_wait_asynccnt)
#define USE_ASYNC_LDS 1
#else
#define USE_ASYNC_LDS 0
#endif

__device__ inline __bf16 to_bf(float x) {
    union { float f; unsigned u; } v; v.f = x;
    unsigned r = (v.u + 0x7FFFu + ((v.u >> 16) & 1u)) >> 16;
    unsigned short h = (unsigned short)r;
    __bf16 out; __builtin_memcpy(&out, &h, 2);
    return out;
}

__device__ inline v8f wmma_bf16(v16bf a, v16bf b, v8f c) {
    return __builtin_amdgcn_wmma_f32_16x16x32_bf16(false, a, false, b, (short)0, c, false, false);
}

// 16-byte global -> LDS copy. Async on CDNA5 (tracked by ASYNCcnt), sync fallback.
// Builtin signature: (int4 __device__* src, int4 __shared__* dst, imm offset, imm cpol);
// HIP device compilation implicitly converts generic pointers to those address spaces.
__device__ inline void cp16_g2l(const __bf16* g, __bf16* l) {
#if USE_ASYNC_LDS
    __builtin_amdgcn_global_load_async_to_lds_b128((vint4*)g, (vint4*)l, 0, 0);
#else
    *(v8bf*)l = *(const v8bf*)g;
#endif
}

// Load a 16x32 bf16 A/B fragment (wave32). Row-major, K contiguous, row stride ldk.
// lane L: row = row0 + (L&15) (clamped), K chunks at klo..klo+7 and klo+16..klo+23.
__device__ inline v16bf load_frag_rows(const __bf16* base, int ldk, int row0, int rowMax) {
    int lane = threadIdx.x & 31;
    int row = row0 + (lane & 15);
    if (row > rowMax) row = rowMax;
    int klo = (lane & 16) >> 1;           // 0 or 8
    const __bf16* p = base + (long)row * ldk + klo;
    v8bf c0 = *(const v8bf*)(p);
    v8bf c1 = *(const v8bf*)(p + 16);
    v16bf r;
#pragma unroll
    for (int e = 0; e < 8; ++e) { r[e] = c0[e]; r[e + 8] = c1[e]; }
    return r;
}

// ---------------- constants ----------------
#define NB    16          // batch
#define NTOK  1025        // tokens per image
#define DM    256         // d_model
#define NH    4           // heads
#define DH    64          // head dim
#define HD    1024        // mlp hidden
#define ROWS  (NB*NTOK)   // 16400
#define VLD   1056        // padded token-stride for vT (33*32)

// ---------------- small prep kernels ----------------
__global__ void rope_tables_kernel(float* cosT, float* sinT) {
    int i = blockIdx.x * blockDim.x + threadIdx.x;
    if (i >= NTOK * 128) return;
    int pos = i >> 7, f = i & 127;
    float invf = __expf(-((float)(2 * f) / 256.0f) * 9.2103403719761836f); // ln(1e4)
    float ang = (float)pos * invf;
    cosT[i] = cosf(ang);
    sinT[i] = sinf(ang);
}

__global__ void cvt_bf16_kernel(const float* __restrict__ src, __bf16* __restrict__ dst, int n) {
    int i = blockIdx.x * blockDim.x + threadIdx.x;
    if (i < n) dst[i] = to_bf(src[i]);
}

// Wt[n*K + k] = W[k*N + n]  (bf16)
__global__ void transpose_bf16_kernel(const float* __restrict__ W, __bf16* __restrict__ Wt, int K, int N) {
    int i = blockIdx.x * blockDim.x + threadIdx.x;
    if (i >= K * N) return;
    int n = i / K, k = i - n * K;
    Wt[i] = to_bf(W[(long)k * N + n]);
}

// A[r, k] gather from x: r = b*1024 + h*32 + w, k = c*256 + p*16 + q
__global__ void patch_gather_kernel(const float* __restrict__ x, __bf16* __restrict__ A) {
    int i = blockIdx.x * blockDim.x + threadIdx.x;
    if (i >= 16384 * 768) return;
    int r = i / 768, k = i - r * 768;
    int b = r >> 10, hw = r & 1023, hh = hw >> 5, ww = hw & 31;
    int c = k >> 8, rem = k & 255, pp = rem >> 4, qq = rem & 15;
    long xi = (((long)(b * 3 + c) * 512) + hh * 16 + pp) * 512 + ww * 16 + qq;
    A[(long)r * 768 + k] = to_bf(x[xi]);
}

__global__ void clsfill_kernel(const float* __restrict__ cls_tok, float* __restrict__ tokens) {
    int i = blockIdx.x * blockDim.x + threadIdx.x;
    if (i >= NB * DM) return;
    int b = i >> 8, d = i & 255;
    tokens[(long)(b * NTOK) * DM + d] = cls_tok[d];
}

// RoPE on consecutive pairs; scale folded in (0.125 for Q, 1.0 for K). f32 in -> bf16 out.
__global__ void rope_kernel(const float* __restrict__ in, const float* __restrict__ cosT,
                            const float* __restrict__ sinT, __bf16* __restrict__ out, float scale) {
    int i = blockIdx.x * blockDim.x + threadIdx.x;
    if (i >= ROWS * 128) return;
    int row = i >> 7, f = i & 127;
    int pos = row % NTOK;
    float c = cosT[pos * 128 + f], s = sinT[pos * 128 + f];
    long base = (long)row * DM + 2 * f;
    float x0 = in[base], x1 = in[base + 1];
    out[base]     = to_bf((x0 * c - x1 * s) * scale);
    out[base + 1] = to_bf((x0 * s + x1 * c) * scale);
}

// ---------------- LayerNorm (one wave per row) ----------------
__global__ __launch_bounds__(256) void ln_kernel(const float* __restrict__ h, const float* __restrict__ g,
                                                 const float* __restrict__ bb, __bf16* __restrict__ y, int rows) {
    int row = blockIdx.x * 8 + (threadIdx.x >> 5);
    int lane = threadIdx.x & 31;
    if (row >= rows) return;
    const float4* p = (const float4*)(h + (long)row * DM);
    float4 a0 = p[lane], a1 = p[lane + 32];
    float s  = a0.x + a0.y + a0.z + a0.w + a1.x + a1.y + a1.z + a1.w;
    float ss = a0.x * a0.x + a0.y * a0.y + a0.z * a0.z + a0.w * a0.w
             + a1.x * a1.x + a1.y * a1.y + a1.z * a1.z + a1.w * a1.w;
#pragma unroll
    for (int m = 16; m >= 1; m >>= 1) { s += __shfl_xor(s, m, 32); ss += __shfl_xor(ss, m, 32); }
    float mean = s * (1.0f / 256.0f);
    float var  = ss * (1.0f / 256.0f) - mean * mean;
    float rstd = rsqrtf(var + 1e-5f);
    __bf16* yo = y + (long)row * DM;
    int c0 = lane * 4;
#pragma unroll
    for (int j = 0; j < 4; ++j) {
        int col = c0 + j; float xv = (&a0.x)[j];
        yo[col] = to_bf((xv - mean) * rstd * g[col] + bb[col]);
    }
#pragma unroll
    for (int j = 0; j < 4; ++j) {
        int col = 128 + c0 + j; float xv = (&a1.x)[j];
        yo[col] = to_bf((xv - mean) * rstd * g[col] + bb[col]);
    }
}

// Final LN over cls rows only -> f32 (grid = NB, block = 32)
__global__ void lnf_kernel(const float* __restrict__ tokens, const float* __restrict__ g,
                           const float* __restrict__ bb, float* __restrict__ clsf) {
    int b = blockIdx.x, lane = threadIdx.x;
    const float4* p = (const float4*)(tokens + (long)(b * NTOK) * DM);
    float4 a0 = p[lane], a1 = p[lane + 32];
    float s  = a0.x + a0.y + a0.z + a0.w + a1.x + a1.y + a1.z + a1.w;
    float ss = a0.x * a0.x + a0.y * a0.y + a0.z * a0.z + a0.w * a0.w
             + a1.x * a1.x + a1.y * a1.y + a1.z * a1.z + a1.w * a1.w;
#pragma unroll
    for (int m = 16; m >= 1; m >>= 1) { s += __shfl_xor(s, m, 32); ss += __shfl_xor(ss, m, 32); }
    float mean = s * (1.0f / 256.0f);
    float rstd = rsqrtf(ss * (1.0f / 256.0f) - mean * mean + 1e-5f);
    int c0 = lane * 4;
#pragma unroll
    for (int j = 0; j < 4; ++j) {
        int col = c0 + j;
        clsf[b * DM + col] = ((&a0.x)[j] - mean) * rstd * g[col] + bb[col];
        col += 128;
        clsf[b * DM + col] = ((&a1.x)[j] - mean) * rstd * g[col] + bb[col];
    }
}

// Classifier head: one thread per (b, class), 256-wide dot (tiny: 16x1000x256)
__global__ void head_kernel(const float* __restrict__ clsf, const float* __restrict__ hw,
                            const float* __restrict__ hb, float* __restrict__ out) {
    int i = blockIdx.x * blockDim.x + threadIdx.x;
    if (i >= NB * 1000) return;
    int b = i / 1000, n = i - b * 1000;
    float acc = hb[n];
    const float* c = clsf + b * DM;
#pragma unroll 8
    for (int k = 0; k < DM; ++k) acc = fmaf(c[k], hw[(long)k * 1000 + n], acc);
    out[i] = acc;
}

// ---------------- Generic WMMA GEMM: C(MxN) = A(MxK bf16) x Bt(N rows x K bf16) ----------------
// Block 128 threads = 4 waves, 64x64 C tile; wave w owns rows w*16..w*16+15, all 64 cols.
// Double-buffered: async global->LDS copies for tile t+1 overlap WMMAs on tile t.
// Each wave issues 4 async-load instructions per tile, so s_wait_asynccnt(4)
// retires exactly the previous tile (async loads complete in order).
// Epilogues: 0=patch(row remap+bias) 1=bias->f32 2=V scatter to (b,h,dh,pos) bf16
//            3=bias+residual->f32 4=bias+GELU->bf16 5=bias+residual->f32
template <int EPI>
__global__ __launch_bounds__(128) void gemm_bf16_kernel(
        const __bf16* __restrict__ A, const __bf16* __restrict__ Bt, const float* __restrict__ bias,
        float* __restrict__ outF, __bf16* __restrict__ outB, const float* __restrict__ resid,
        int M, int N, int K) {
    __shared__ __align__(16) __bf16 Al[2][64 * 32];
    __shared__ __align__(16) __bf16 Bl[2][64 * 32];
    int tid = threadIdx.x;
    int m0 = blockIdx.x * 64, n0 = blockIdx.y * 64;
    int wv = tid >> 5, lane = tid & 31;
    v8f acc[4];
#pragma unroll
    for (int f = 0; f < 4; ++f) acc[f] = (v8f){0, 0, 0, 0, 0, 0, 0, 0};

    auto issue = [&](int buf, int k0) {
#pragma unroll
        for (int i = tid; i < 256; i += 128) {          // 64 rows x 4 chunks of 8 bf16
            int rr = i >> 2, c8 = (i & 3) * 8;
            int ga = m0 + rr; if (ga > M - 1) ga = M - 1;
            cp16_g2l(A  + (long)ga * K + k0 + c8,        Al[buf] + rr * 32 + c8);
            cp16_g2l(Bt + (long)(n0 + rr) * K + k0 + c8, Bl[buf] + rr * 32 + c8);
        }
    };

    int nk = K >> 5;
    issue(0, 0);
    for (int t = 0; t < nk; ++t) {
        int cur = t & 1;
        if (t + 1 < nk) {
            issue(cur ^ 1, (t + 1) << 5);
#if USE_ASYNC_LDS
            __builtin_amdgcn_s_wait_asynccnt(4);   // tile t's 4 copies done; t+1 in flight
#endif
        } else {
#if USE_ASYNC_LDS
            __builtin_amdgcn_s_wait_asynccnt(0);
#endif
        }
        __syncthreads();                            // cross-wave visibility of LDS tile
        v16bf a = load_frag_rows(Al[cur] + wv * 16 * 32, 32, 0, 15);
#pragma unroll
        for (int f = 0; f < 4; ++f) {
            v16bf b = load_frag_rows(Bl[cur] + f * 16 * 32, 32, 0, 15);
            acc[f] = wmma_bf16(a, b, acc[f]);
        }
        __syncthreads();                            // all reads done before buffer reuse
    }

    int nn = lane & 15, mb = (lane >> 4) * 8;
#pragma unroll
    for (int f = 0; f < 4; ++f) {
        int col = n0 + f * 16 + nn;
#pragma unroll
        for (int r = 0; r < 8; ++r) {
            int row = m0 + wv * 16 + mb + r;
            if (row >= M) continue;
            float val = acc[f][r] + bias[col];
            if constexpr (EPI == 0) {               // patch embed -> tokens (skip cls slot)
                int tr = (row >> 10) * NTOK + 1 + (row & 1023);
                outF[(long)tr * DM + col] = val;
            } else if constexpr (EPI == 1) {        // plain bias, f32 (pre-RoPE q/k)
                outF[(long)row * N + col] = val;
            } else if constexpr (EPI == 2) {        // V: scatter to (b,h,dh,pos) bf16
                int b2 = row / NTOK, pos = row - b2 * NTOK;
                int hh = col >> 6, dc = col & 63;
                outB[((long)((b2 * NH + hh) * DH + dc)) * VLD + pos] = to_bf(val);
            } else if constexpr (EPI == 3) {        // O-proj + residual
                outF[(long)row * N + col] = val + resid[(long)row * N + col];
            } else if constexpr (EPI == 4) {        // MLP1: exact GELU -> bf16
                float gl = 0.5f * val * (1.0f + erff(val * 0.70710678118654752f));
                outB[(long)row * N + col] = to_bf(gl);
            } else {                                // MLP2 + residual
                outF[(long)row * N + col] = val + resid[(long)row * N + col];
            }
        }
    }
}

// ---------------- Flash attention ----------------
// Block 128 = 4 waves; wave handles one (b, h, qtile-of-16). 65 q-tiles, 33 iters of 32 keys.
// All LDS traffic is intra-wave (DS ops in order per wave) -> no barriers needed.
__global__ __launch_bounds__(128) void attn_kernel(const __bf16* __restrict__ q, const __bf16* __restrict__ k,
                                                   const __bf16* __restrict__ vT, __bf16* __restrict__ ctx) {
    __shared__ __align__(16) float  Sl[4][16 * 32];
    __shared__ __align__(16) __bf16 Pl[4][16 * 32];
    __shared__ float corrL[4][16];
    __shared__ float invL[4][16];

    int wv = threadIdx.x >> 5, lane = threadIdx.x & 31;
    int g = blockIdx.x * 4 + wv;                 // 0..4159
    int b = g / (NH * 65); int rem = g - b * (NH * 65);
    int h = rem / 65; int qt = rem - h * 65;

    float*  S    = Sl[wv];
    __bf16* P    = Pl[wv];
    float*  corr = corrL[wv];
    float*  inv  = invL[wv];
    int nn = lane & 15, mb = (lane >> 4) * 8;

    const __bf16* qbase = q + (long)(b * NTOK) * DM + h * DH;
    v16bf q0 = load_frag_rows(qbase,      DM, qt * 16, NTOK - 1);   // dh 0..31
    v16bf q1 = load_frag_rows(qbase + 32, DM, qt * 16, NTOK - 1);   // dh 32..63

    v8f o[4];
#pragma unroll
    for (int f = 0; f < 4; ++f) o[f] = (v8f){0, 0, 0, 0, 0, 0, 0, 0};
    float m_i = -1e30f, l_i = 0.0f;              // per-row stats, lanes 0..15 own row==lane

    const __bf16* kbase = k + (long)(b * NTOK) * DM + h * DH;
    const __bf16* vbase = vT + (long)((b * NH + h) * DH) * VLD;

    for (int kt = 0; kt < 33; ++kt) {
        int kr0 = kt * 32;

        // prefetch K rows two tiles ahead (global_prefetch_b8; streams from L2/HBM)
        int pfrow = kr0 + 64 + (lane & 15) * 2;
        if (pfrow < NTOK) __builtin_prefetch(kbase + (long)pfrow * DM, 0, 0);

        v16bf kb00 = load_frag_rows(kbase,      DM, kr0,      NTOK - 1);
        v16bf kb01 = load_frag_rows(kbase + 32, DM, kr0,      NTOK - 1);
        v16bf kb10 = load_frag_rows(kbase,      DM, kr0 + 16, NTOK - 1);
        v16bf kb11 = load_frag_rows(kbase + 32, DM, kr0 + 16, NTOK - 1);
        v8f s0 = (v8f){0, 0, 0, 0, 0, 0, 0, 0};
        v8f s1 = (v8f){0, 0, 0, 0, 0, 0, 0, 0};
        s0 = wmma_bf16(q0, kb00, s0); s0 = wmma_bf16(q1, kb01, s0);
        s1 = wmma_bf16(q0, kb10, s1); s1 = wmma_bf16(q1, kb11, s1);

        int kt0 = kr0 + nn, kt1 = kr0 + 16 + nn;
#pragma unroll
        for (int r = 0; r < 8; ++r) {            // softmax scale already folded into Q
            S[(mb + r) * 32 + nn]      = (kt0 < NTOK) ? s0[r] : -1e30f;
            S[(mb + r) * 32 + 16 + nn] = (kt1 < NTOK) ? s1[r] : -1e30f;
        }

        if (lane < 16) {                         // streaming softmax for row == lane
            int rr = lane;
            float mx = m_i;
#pragma unroll
            for (int j = 0; j < 32; ++j) mx = fmaxf(mx, S[rr * 32 + j]);
            float cf = __expf(m_i - mx);
            float sum = 0.0f;
#pragma unroll
            for (int j = 0; j < 32; ++j) {
                float pv = __expf(S[rr * 32 + j] - mx);
                P[rr * 32 + j] = to_bf(pv);
                sum += pv;
            }
            l_i = l_i * cf + sum;
            m_i = mx;
            corr[rr] = cf;
        }

        float cr[8];
#pragma unroll
        for (int r = 0; r < 8; ++r) cr[r] = corr[mb + r];
#pragma unroll
        for (int f = 0; f < 4; ++f)
#pragma unroll
            for (int r = 0; r < 8; ++r) o[f][r] *= cr[r];

        v16bf pf = load_frag_rows(P, 32, 0, 15);
#pragma unroll
        for (int f = 0; f < 4; ++f) {
            v16bf vf = load_frag_rows(vbase + (long)(f * 16) * VLD + kr0, VLD, 0, 15);
            o[f] = wmma_bf16(pf, vf, o[f]);
        }
    }

    if (lane < 16) inv[lane] = 1.0f / l_i;
    float ivr[8];
#pragma unroll
    for (int r = 0; r < 8; ++r) ivr[r] = inv[mb + r];
#pragma unroll
    for (int f = 0; f < 4; ++f)
#pragma unroll
        for (int r = 0; r < 8; ++r) {
            int tok = qt * 16 + mb + r;
            if (tok < NTOK)
                ctx[(long)(b * NTOK + tok) * DM + h * DH + f * 16 + nn] = to_bf(o[f][r] * ivr[r]);
        }
}

// ---------------- host ----------------
static inline int cdiv(int a, int b) { return (a + b - 1) / b; }

extern "C" void kernel_launch(void* const* d_in, const int* in_sizes, int n_in,
                              void* d_out, int out_size, void* d_ws, size_t ws_size,
                              hipStream_t stream) {
    const float* x       = (const float*)d_in[0];
    const float* conv_w  = (const float*)d_in[1];
    const float* conv_b  = (const float*)d_in[2];
    const float* cls_tok = (const float*)d_in[3];
    const float* ln1_g   = (const float*)d_in[4];
    const float* ln1_b   = (const float*)d_in[5];
    const float* wq      = (const float*)d_in[6];
    const float* bq      = (const float*)d_in[7];
    const float* wk      = (const float*)d_in[8];
    const float* bk      = (const float*)d_in[9];
    const float* wv      = (const float*)d_in[10];
    const float* bv      = (const float*)d_in[11];
    const float* wo      = (const float*)d_in[12];
    const float* bo      = (const float*)d_in[13];
    const float* ln2_g   = (const float*)d_in[14];
    const float* ln2_b   = (const float*)d_in[15];
    const float* w1      = (const float*)d_in[16];
    const float* b1      = (const float*)d_in[17];
    const float* w2      = (const float*)d_in[18];
    const float* b2      = (const float*)d_in[19];
    const float* lnf_g   = (const float*)d_in[20];
    const float* lnf_b   = (const float*)d_in[21];
    const float* head_w  = (const float*)d_in[22];
    const float* head_b  = (const float*)d_in[23];
    float* out = (float*)d_out;

    char* p = (char*)d_ws;
    auto alloc = [&](size_t bytes) -> char* {
        char* r = p; p += (bytes + 255) & ~(size_t)255; return r;
    };
    float*  tokens = (float*) alloc((size_t)ROWS * DM * 4);       // residual stream (f32)
    __bf16* ybf    = (__bf16*)alloc((size_t)ROWS * DM * 2);       // LN output (bf16 GEMM A)
    float*  tmpf   = (float*) alloc((size_t)ROWS * DM * 4);       // q/k pre-RoPE
    __bf16* qb     = (__bf16*)alloc((size_t)ROWS * DM * 2);
    __bf16* kb     = (__bf16*)alloc((size_t)ROWS * DM * 2);
    __bf16* vT     = (__bf16*)alloc((size_t)NB * NH * DH * VLD * 2);
    __bf16* ctx    = (__bf16*)alloc((size_t)ROWS * DM * 2);
    __bf16* hid    = (__bf16*)alloc((size_t)ROWS * HD * 2);
    __bf16* patA   = (__bf16*)alloc((size_t)16384 * 768 * 2);
    __bf16* cwbf   = (__bf16*)alloc((size_t)256 * 768 * 2);
    __bf16* wT     = (__bf16*)alloc((size_t)8 * 786432 * 2);      // per-layer transposed weights
    float*  cosT   = (float*) alloc((size_t)NTOK * 128 * 4);
    float*  sinT   = (float*) alloc((size_t)NTOK * 128 * 4);
    float*  clsf   = (float*) alloc((size_t)NB * DM * 4);
    (void)ws_size; (void)n_in; (void)in_sizes; (void)out_size;

    // ---- prep ----
    rope_tables_kernel<<<cdiv(NTOK * 128, 256), 256, 0, stream>>>(cosT, sinT);
    cvt_bf16_kernel<<<cdiv(256 * 768, 256), 256, 0, stream>>>(conv_w, cwbf, 256 * 768);
    for (int l = 0; l < 8; ++l) {
        __bf16* wb = wT + (size_t)l * 786432;
        transpose_bf16_kernel<<<cdiv(65536, 256), 256, 0, stream>>>(wq + (size_t)l * 65536, wb,            256, 256);
        transpose_bf16_kernel<<<cdiv(65536, 256), 256, 0, stream>>>(wk + (size_t)l * 65536, wb + 65536,    256, 256);
        transpose_bf16_kernel<<<cdiv(65536, 256), 256, 0, stream>>>(wv + (size_t)l * 65536, wb + 131072,   256, 256);
        transpose_bf16_kernel<<<cdiv(65536, 256), 256, 0, stream>>>(wo + (size_t)l * 65536, wb + 196608,   256, 256);
        transpose_bf16_kernel<<<cdiv(262144, 256), 256, 0, stream>>>(w1 + (size_t)l * 262144, wb + 262144, 256, 1024);
        transpose_bf16_kernel<<<cdiv(262144, 256), 256, 0, stream>>>(w2 + (size_t)l * 262144, wb + 524288, 1024, 256);
    }
    patch_gather_kernel<<<cdiv(16384 * 768, 256), 256, 0, stream>>>(x, patA);

    // ---- patch embed GEMM: (16384 x 768) x (768 x 256) -> tokens rows 1..1024 per image ----
    gemm_bf16_kernel<0><<<dim3(16384 / 64, 256 / 64), 128, 0, stream>>>(
        patA, cwbf, conv_b, tokens, nullptr, nullptr, 16384, 256, 768);
    clsfill_kernel<<<cdiv(NB * DM, 256), 256, 0, stream>>>(cls_tok, tokens);

    dim3 g64(cdiv(ROWS, 64), 256 / 64);
    dim3 g64h(cdiv(ROWS, 64), HD / 64);

    // ---- transformer layers ----
    for (int l = 0; l < 8; ++l) {
        const __bf16* wqT = wT + (size_t)l * 786432;
        const __bf16* wkT = wqT + 65536;
        const __bf16* wvT = wqT + 131072;
        const __bf16* woT = wqT + 196608;
        const __bf16* w1T = wqT + 262144;
        const __bf16* w2T = wqT + 524288;

        ln_kernel<<<cdiv(ROWS, 8), 256, 0, stream>>>(tokens, ln1_g + l * 256, ln1_b + l * 256, ybf, ROWS);

        gemm_bf16_kernel<1><<<g64, 128, 0, stream>>>(ybf, wqT, bq + l * 256, tmpf, nullptr, nullptr, ROWS, 256, 256);
        rope_kernel<<<cdiv(ROWS * 128, 256), 256, 0, stream>>>(tmpf, cosT, sinT, qb, 0.125f); // 1/sqrt(64)

        gemm_bf16_kernel<1><<<g64, 128, 0, stream>>>(ybf, wkT, bk + l * 256, tmpf, nullptr, nullptr, ROWS, 256, 256);
        rope_kernel<<<cdiv(ROWS * 128, 256), 256, 0, stream>>>(tmpf, cosT, sinT, kb, 1.0f);

        gemm_bf16_kernel<2><<<g64, 128, 0, stream>>>(ybf, wvT, bv + l * 256, nullptr, vT, nullptr, ROWS, 256, 256);

        attn_kernel<<<NB * NH * 65 / 4, 128, 0, stream>>>(qb, kb, vT, ctx);

        gemm_bf16_kernel<3><<<g64, 128, 0, stream>>>(ctx, woT, bo + l * 256, tokens, nullptr, tokens, ROWS, 256, 256);

        ln_kernel<<<cdiv(ROWS, 8), 256, 0, stream>>>(tokens, ln2_g + l * 256, ln2_b + l * 256, ybf, ROWS);

        gemm_bf16_kernel<4><<<g64h, 128, 0, stream>>>(ybf, w1T, b1 + l * 1024, nullptr, hid, nullptr, ROWS, 1024, 256);
        gemm_bf16_kernel<5><<<g64, 128, 0, stream>>>(hid, w2T, b2 + l * 256, tokens, nullptr, tokens, ROWS, 256, 1024);
    }

    // ---- final LN (cls rows) + head ----
    lnf_kernel<<<NB, 32, 0, stream>>>(tokens, lnf_g, lnf_b, clsf);
    head_kernel<<<cdiv(NB * 1000, 256), 256, 0, stream>>>(clsf, head_w, head_b, out);
}
